// Label_Self_Attention_28621662060729
// MI455X (gfx1250) — compile-verified, matrix operand order checked by reading, not run on hardware
//
#include <hip/hip_runtime.h>
#include <hip/hip_bf16.h>

typedef __bf16 bf16;
typedef __attribute__((ext_vector_type(16))) __bf16 v16bf;
typedef __attribute__((ext_vector_type(8)))  __bf16 v8bf;
typedef __attribute__((ext_vector_type(8)))  float  v8f;

#define NB   8
#define CIN  256
#define HH   128
#define WW   128
#define HWT  16384
#define AD   64
#define K2   289
#define K2P  320
#define W2   17
#define LDP  72    // padded K-contiguous leading dim for 64-deep K tiles

static __device__ __forceinline__ v8f wmma_bf16(v16bf a, v16bf b, v8f c) {
  return __builtin_amdgcn_wmma_f32_16x16x32_bf16(false, a, false, b, (short)0, c, false, false);
}

// Unified fragment loader. Operand stored [row][K] with K contiguous, ld multiple of 8.
// For A: row = M index. For B: row = N index (i.e. B stored transposed).
// Lanes 0-15: row r0+lane, K base k0; lanes 16-31: row r0+lane-16, K base k0+8.
// 16 bf16 = two contiguous 16-byte runs -> two ds_load_b128.
static __device__ __forceinline__ v16bf load_frag(const bf16* P, int ld, int r0, int k0, int lane) {
  const bf16* p = P + (r0 + (lane & 15)) * ld + k0 + ((lane >> 4) << 3);
  v8bf lo = *(const v8bf*)(p);
  v8bf hi = *(const v8bf*)(p + 16);
  return __builtin_shufflevector(lo, hi, 0, 1, 2, 3, 4, 5, 6, 7, 8, 9, 10, 11, 12, 13, 14, 15);
}

static __device__ __forceinline__ unsigned lds_lo32(const void* p) {
  return (unsigned)(size_t)p;
}
// CDNA5 async global->LDS copy of 16 bytes (GVS mode), tracked by ASYNCcnt.
static __device__ __forceinline__ void async_copy16(unsigned ldsAddr, const void* gbase, int byteOff) {
  asm volatile("global_load_async_to_lds_b128 %0, %1, %2"
               :: "v"(ldsAddr + byteOff), "v"(byteOff), "s"(gbase)
               : "memory");
}
static __device__ __forceinline__ void async_wait0() {
  asm volatile("s_wait_asynccnt 0x0" ::: "memory");
}

// order-preserving float<->uint mapping for atomic max
static __device__ __forceinline__ unsigned fmap(float f) {
  unsigned u = __float_as_uint(f);
  return (u & 0x80000000u) ? ~u : (u | 0x80000000u);
}
static __device__ __forceinline__ float funmap(unsigned u) {
  return (u & 0x80000000u) ? __uint_as_float(u & 0x7fffffffu) : __uint_as_float(~u);
}

static __device__ __forceinline__ unsigned pack_bf16(float a, float b) {
  bf16 h0 = (bf16)a, h1 = (bf16)b;
  unsigned short u0, u1;
  __builtin_memcpy(&u0, &h0, 2);
  __builtin_memcpy(&u1, &h1, 2);
  return (unsigned)u0 | ((unsigned)u1 << 16);
}

// ---------------------------------------------------------------------------
// P1: downsample x (bilinear align_corners) -> k^T = (wk*xd + bk)^T  (K2P x A)
//     downsample logits -> v  (A x K2P). Pad rows/cols [289,320) with zeros.
__global__ void __launch_bounds__(256) p1_prep(const float* __restrict__ x,
                                               const float* __restrict__ logits,
                                               const float* __restrict__ wk,
                                               const float* __restrict__ bk,
                                               bf16* __restrict__ kTg,
                                               bf16* __restrict__ vg) {
  __shared__ float xdL[CIN];
  int n  = blockIdx.x / K2P;
  int k2 = blockIdx.x % K2P;
  int t  = threadIdx.x;
  if (k2 >= K2) {
    if (t < AD) {
      kTg[((size_t)n * K2P + k2) * AD + t] = (bf16)0.0f;
      vg[((size_t)n * AD + t) * K2P + k2]  = (bf16)0.0f;
    }
    return;
  }
  int   jy = k2 / W2, jx = k2 % W2;
  float ys = jy * (127.0f / 16.0f), xs = jx * (127.0f / 16.0f);
  int   y0 = (int)ys, x0 = (int)xs;
  int   y1 = min(y0 + 1, HH - 1), x1 = min(x0 + 1, WW - 1);
  float fy = ys - (float)y0, fx = xs - (float)x0;
  {
    const float* p = x + ((size_t)(n * CIN + t)) * HWT;
    float v00 = p[y0 * WW + x0], v01 = p[y0 * WW + x1];
    float v10 = p[y1 * WW + x0], v11 = p[y1 * WW + x1];
    xdL[t] = (v00 * (1.f - fx) + v01 * fx) * (1.f - fy) + (v10 * (1.f - fx) + v11 * fx) * fy;
  }
  __syncthreads();
  if (t < AD) {
    const float* p = logits + ((size_t)(n * AD + t)) * HWT;
    float v00 = p[y0 * WW + x0], v01 = p[y0 * WW + x1];
    float v10 = p[y1 * WW + x0], v11 = p[y1 * WW + x1];
    float vv = (v00 * (1.f - fx) + v01 * fx) * (1.f - fy) + (v10 * (1.f - fx) + v11 * fx) * fy;
    vg[((size_t)n * AD + t) * K2P + k2] = (bf16)vv;
    float        s  = bk[t];
    const float* wr = wk + t * CIN;
#pragma unroll 8
    for (int c = 0; c < CIN; ++c) s += wr[c] * xdL[c];
    kTg[((size_t)n * K2P + k2) * AD + t] = (bf16)s;
  }
}

// ---------------------------------------------------------------------------
// P2: q = wq * x + bq  (A x HW per batch), bf16 output. WMMA M=64 K=256 N=16384.
// x slab staged transposed [s][c] so B-fragments are contiguous b128 loads.
__global__ void __launch_bounds__(256) p2_q(const float* __restrict__ x,
                                            const float* __restrict__ wq,
                                            const float* __restrict__ bq,
                                            bf16* __restrict__ qg) {
  extern __shared__ char sm[];
  bf16* wqL = (bf16*)sm;                        // 64 x 256 (row-major, A operand)
  bf16* xsT = (bf16*)(sm + AD * CIN * 2);       // 256 x LDP transposed slab
  int n = blockIdx.x >> 6;
  int ch = blockIdx.x & 63;
  int s0 = ch * 256;
  int t = threadIdx.x, wave = t >> 5, lane = t & 31;
  for (int i = t; i < AD * CIN; i += 256) wqL[i] = (bf16)wq[i];

  v8f acc[4][2];
#pragma unroll
  for (int m = 0; m < 4; ++m)
#pragma unroll
    for (int q2 = 0; q2 < 2; ++q2)
#pragma unroll
      for (int e = 0; e < 8; ++e) acc[m][q2][e] = 0.0f;

  for (int kc = 0; kc < 4; ++kc) {
    __syncthreads();
    for (int r = 0; r < 64; ++r)
      xsT[t * LDP + r] = (bf16)x[((size_t)(n * CIN + kc * 64 + r)) * HWT + s0 + t];
    __syncthreads();
#pragma unroll
    for (int ks = 0; ks < 2; ++ks) {
      int   kg = kc * 64 + ks * 32;
      v16bf b0 = load_frag(xsT, LDP, wave * 32, ks * 32, lane);
      v16bf b1 = load_frag(xsT, LDP, wave * 32 + 16, ks * 32, lane);
#pragma unroll
      for (int m = 0; m < 4; ++m) {
        v16bf a = load_frag(wqL, CIN, m * 16, kg, lane);
        acc[m][0] = wmma_bf16(a, b0, acc[m][0]);
        acc[m][1] = wmma_bf16(a, b1, acc[m][1]);
      }
    }
  }
  int col = lane & 15, hi = (lane >> 4) << 3;
#pragma unroll
  for (int m = 0; m < 4; ++m)
#pragma unroll
    for (int q2 = 0; q2 < 2; ++q2)
#pragma unroll
      for (int v = 0; v < 8; ++v) {
        int a  = m * 16 + hi + v;
        int ss = s0 + wave * 32 + q2 * 16 + col;
        qg[((size_t)(n * AD + a)) * HWT + ss] = (bf16)(acc[m][q2][v] + bq[a]);
      }
}

// ---------------------------------------------------------------------------
// P3: per-(n,k2-row) softmax stats over a 1024-wide s-chunk. S = kT*q * 0.125.
__global__ void __launch_bounds__(256) p3_stats(const bf16* __restrict__ kTg,
                                                const bf16* __restrict__ qg,
                                                float* __restrict__ pmax,
                                                float* __restrict__ psum) {
  extern __shared__ char sm[];
  bf16*     kTL   = (bf16*)sm;                                   // 320 x 64
  bf16*     qLT   = (bf16*)(sm + K2P * AD * 2);                  // 128 x LDP (transposed)
  unsigned* umaxL = (unsigned*)(sm + K2P * AD * 2 + 128 * LDP * 2);
  float*    fmaxL = (float*)((char*)umaxL + K2P * 4);
  float*    fsumL = (float*)((char*)fmaxL + K2P * 4);
  int n = blockIdx.x >> 4, chunk = blockIdx.x & 15;
  int t = threadIdx.x, wave = t >> 5, lane = t & 31;
  int hi = (lane >> 4) << 3;

  // async copy kT (pure bf16 copy) into LDS
  {
    const void* gsrc = (const void*)(kTg + (size_t)n * K2P * AD);
    unsigned    ldst = lds_lo32(kTL);
    for (int i = t; i < (K2P * AD * 2) / 16; i += 256) async_copy16(ldst, gsrc, i * 16);
  }
  for (int i = t; i < K2P; i += 256) { umaxL[i] = 0u; fsumL[i] = 0.0f; }
  async_wait0();
  int sbase = chunk * 1024;

  for (int pass = 0; pass < 2; ++pass) {
    for (int it = 0; it < 8; ++it) {
      __syncthreads();
      {
        int col = t & 127, rb = (t >> 7) * 32;
        for (int r = 0; r < 32; ++r)
          qLT[col * LDP + rb + r] =
              qg[((size_t)(n * AD + rb + r)) * HWT + sbase + it * 128 + col];
      }
      __syncthreads();
      for (int m = 0; m < 20; ++m) {
        v8f acc;
#pragma unroll
        for (int e = 0; e < 8; ++e) acc[e] = 0.0f;
#pragma unroll
        for (int ks = 0; ks < 2; ++ks)
          acc = wmma_bf16(load_frag(kTL, AD, m * 16, ks * 32, lane),
                          load_frag(qLT, LDP, wave * 16, ks * 32, lane), acc);
#pragma unroll
        for (int v = 0; v < 8; ++v) {
          int   row = m * 16 + hi + v;
          float sc  = 0.125f * acc[v];
          if (pass == 0)
            atomicMax(&umaxL[row], fmap(sc));
          else
            atomicAdd(&fsumL[row], __expf(sc - fmaxL[row]));
        }
      }
    }
    __syncthreads();
    if (pass == 0) {
      for (int i = t; i < K2P; i += 256) fmaxL[i] = funmap(umaxL[i]);
      __syncthreads();
    }
  }
  for (int i = t; i < K2P; i += 256) {
    pmax[((size_t)(n * 16 + chunk)) * K2P + i] = fmaxL[i];
    psum[((size_t)(n * 16 + chunk)) * K2P + i] = fsumL[i];
  }
}

// P3b: combine the 16 chunk partials into final rowmax + 1/rowsum.
__global__ void p3b_combine(const float* __restrict__ pmax, const float* __restrict__ psum,
                            float* __restrict__ smax, float* __restrict__ sinv) {
  int tid = blockIdx.x * 256 + threadIdx.x;
  if (tid >= NB * K2P) return;
  int   n = tid / K2P, row = tid % K2P;
  float m = -1e30f;
  for (int c = 0; c < 16; ++c) m = fmaxf(m, pmax[((size_t)(n * 16 + c)) * K2P + row]);
  float s = 0.0f;
  for (int c = 0; c < 16; ++c)
    s += psum[((size_t)(n * 16 + c)) * K2P + row] *
         __expf(pmax[((size_t)(n * 16 + c)) * K2P + row] - m);
  smax[tid] = m;
  sinv[tid] = (s > 0.0f) ? 1.0f / s : 0.0f;
}

// ---------------------------------------------------------------------------
// P4: fully fused per (n, h): att tile in LDS -> v*att + score -> wo*z + bo ->
//     softmax over W -> out. att never touches HBM.
__global__ void __launch_bounds__(256) p4_fused(const bf16* __restrict__ kTg,
                                                const bf16* __restrict__ vg,
                                                const bf16* __restrict__ qg,
                                                const float* __restrict__ logits,
                                                const float* __restrict__ wo,
                                                const float* __restrict__ bo,
                                                const float* __restrict__ smax,
                                                const float* __restrict__ sinv,
                                                float* __restrict__ outg) {
  extern __shared__ char sm[];
  size_t off = 0;
  bf16*  kTL  = (bf16*)(sm + off);  off += (size_t)K2P * AD * 2;   // 40960
  bf16*  vL   = (bf16*)(sm + off);  off += (size_t)AD * K2P * 2;   // 40960
  bf16*  woL  = (bf16*)(sm + off);  off += (size_t)AD * AD * 2;    // 8192
  bf16*  qLT  = (bf16*)(sm + off);  off += (size_t)128 * LDP * 2;  // 18432
  bf16*  attT = (bf16*)(sm + off);  off += (size_t)128 * K2P * 2;  // 81920
  float* rmL  = (float*)(sm + off); off += (size_t)K2P * 4;
  float* riL  = (float*)(sm + off); off += (size_t)K2P * 4;
  float* outL = (float*)(sm + off); off += (size_t)AD * 128 * 4;   // 32768
  float* mrow = (float*)(sm + off); off += (size_t)AD * 4;
  float* irow = (float*)(sm + off);

  int n = blockIdx.x >> 7, h = blockIdx.x & 127;
  int t = threadIdx.x, wave = t >> 5, lane = t & 31;
  int hi = (lane >> 4) << 3, col = lane & 15;

  // Async global->LDS copies for the pure bf16/f32 copies (ASYNCcnt path).
  {
    const void* gk = (const void*)(kTg + (size_t)n * K2P * AD);
    const void* gv = (const void*)(vg + (size_t)n * AD * K2P);
    const void* gm = (const void*)(smax + (size_t)n * K2P);
    const void* gi = (const void*)(sinv + (size_t)n * K2P);
    unsigned lk = lds_lo32(kTL), lv = lds_lo32(vL), lm = lds_lo32(rmL), li = lds_lo32(riL);
    for (int i = t; i < (K2P * AD * 2) / 16; i += 256) async_copy16(lk, gk, i * 16);
    for (int i = t; i < (AD * K2P * 2) / 16; i += 256) async_copy16(lv, gv, i * 16);
    for (int i = t; i < (K2P * 4) / 16; i += 256) async_copy16(lm, gm, i * 16);
    for (int i = t; i < (K2P * 4) / 16; i += 256) async_copy16(li, gi, i * 16);
  }
  for (int i = t; i < AD * AD; i += 256) woL[i] = (bf16)wo[i];
  {
    int c2 = t & 127, rb = (t >> 7) * 32;
    for (int r = 0; r < 32; ++r)
      qLT[c2 * LDP + rb + r] = qg[((size_t)(n * AD + rb + r)) * HWT + h * 128 + c2];
  }
  async_wait0();
  __syncthreads();

  // Phase A: att tile (K2P x 128), stored transposed [s][k2] in LDS as bf16.
  for (int m = 0; m < 20; ++m) {
    v8f acc;
#pragma unroll
    for (int e = 0; e < 8; ++e) acc[e] = 0.0f;
#pragma unroll
    for (int ks = 0; ks < 2; ++ks)
      acc = wmma_bf16(load_frag(kTL, AD, m * 16, ks * 32, lane),
                      load_frag(qLT, LDP, wave * 16, ks * 32, lane), acc);
#pragma unroll
    for (int v = 0; v < 4; ++v) {
      int   row = m * 16 + hi + 2 * v;
      float e0  = __expf(0.125f * acc[2 * v] - rmL[row]) * riL[row];
      float e1  = __expf(0.125f * acc[2 * v + 1] - rmL[row + 1]) * riL[row + 1];
      *(unsigned*)((char*)attT + (((size_t)(wave * 16 + col)) * K2P + row) * 2) = pack_bf16(e0, e1);
    }
  }
  __syncthreads();

  // Phase B: z = v * att + score   (64 x 128)
  v8f accB[4];
#pragma unroll
  for (int m = 0; m < 4; ++m)
#pragma unroll
    for (int e = 0; e < 8; ++e) accB[m][e] = 0.0f;
  for (int kc = 0; kc < 10; ++kc) {
    v16bf b = load_frag(attT, K2P, wave * 16, kc * 32, lane);
#pragma unroll
    for (int m = 0; m < 4; ++m)
      accB[m] = wmma_bf16(load_frag(vL, K2P, m * 16, kc * 32, lane), b, accB[m]);
  }
#pragma unroll
  for (int m = 0; m < 4; ++m)
#pragma unroll
    for (int v = 0; v < 8; ++v) {
      int r = m * 16 + hi + v;
      accB[m][v] += logits[(((size_t)n * AD + r) * HH + h) * WW + wave * 16 + col];
    }

  // Phase C: out = wo * z + bo. z accumulators convert to B-fragments in-lane.
  v8f accO[4];
#pragma unroll
  for (int m = 0; m < 4; ++m)
#pragma unroll
    for (int e = 0; e < 8; ++e) accO[m][e] = 0.0f;
#pragma unroll
  for (int ks = 0; ks < 2; ++ks) {
    v16bf zb;
#pragma unroll
    for (int j = 0; j < 8; ++j) {
      int mt = 2 * ks + ((j >= 4) ? 1 : 0);
      int vv = (j < 4) ? 2 * j : 2 * j - 8;
      zb[2 * j]     = (bf16)accB[mt][vv];
      zb[2 * j + 1] = (bf16)accB[mt][vv + 1];
    }
#pragma unroll
    for (int m = 0; m < 4; ++m)
      accO[m] = wmma_bf16(load_frag(woL, AD, m * 16, ks * 32, lane), zb, accO[m]);
  }
#pragma unroll
  for (int m = 0; m < 4; ++m)
#pragma unroll
    for (int v = 0; v < 8; ++v) {
      int r = m * 16 + hi + v;
      outL[r * 128 + wave * 16 + col] = accO[m][v] + bo[r];
    }
  __syncthreads();

  // Phase D: softmax over W (the 128 columns) per channel row.
  if (t < AD) {
    float mx = -1e30f;
    for (int j = 0; j < 128; ++j) mx = fmaxf(mx, outL[t * 128 + j]);
    float s = 0.0f;
    for (int j = 0; j < 128; ++j) s += __expf(outL[t * 128 + j] - mx);
    mrow[t] = mx;
    irow[t] = 1.0f / s;
  }
  __syncthreads();
  for (int i = 0; i < 32; ++i) {
    int e = t + i * 256;
    int r = e >> 7, c2 = e & 127;
    outg[(((size_t)n * AD + r) * HH + h) * WW + c2] = __expf(outL[e] - mrow[r]) * irow[r];
  }
}

// ---------------------------------------------------------------------------
extern "C" void kernel_launch(void* const* d_in, const int* in_sizes, int n_in,
                              void* d_out, int out_size, void* d_ws, size_t ws_size,
                              hipStream_t stream) {
  const float* x      = (const float*)d_in[0];
  const float* logits = (const float*)d_in[1];
  const float* wq     = (const float*)d_in[2];
  const float* bq     = (const float*)d_in[3];
  const float* wk     = (const float*)d_in[4];
  const float* bk     = (const float*)d_in[5];
  const float* wo     = (const float*)d_in[6];
  const float* bo     = (const float*)d_in[7];
  float*       out    = (float*)d_out;

  char*  ws = (char*)d_ws;
  size_t o  = 0;
  bf16*  qg   = (bf16*)(ws + o); o += (size_t)NB * AD * HWT * 2;   // 16 MB
  bf16*  kTg  = (bf16*)(ws + o); o += (size_t)NB * K2P * AD * 2;
  bf16*  vg   = (bf16*)(ws + o); o += (size_t)NB * AD * K2P * 2;
  float* pmax = (float*)(ws + o); o += (size_t)NB * 16 * K2P * 4;
  float* psum = (float*)(ws + o); o += (size_t)NB * 16 * K2P * 4;
  float* smax = (float*)(ws + o); o += (size_t)NB * K2P * 4;
  float* sinv = (float*)(ws + o); o += (size_t)NB * K2P * 4;
  (void)ws_size; (void)in_sizes; (void)n_in; (void)out_size;

  size_t p2_lds = (size_t)AD * CIN * 2 + 256 * LDP * 2;                        // 69632
  size_t p3_lds = (size_t)K2P * AD * 2 + 128 * LDP * 2 + 3 * K2P * 4;          // 63232
  size_t p4_lds = (size_t)K2P * AD * 2 + AD * K2P * 2 + AD * AD * 2 +
                  128 * LDP * 2 + 128 * K2P * 2 + 2 * K2P * 4 + AD * 128 * 4 +
                  2 * AD * 4;                                                  // 226304

  p1_prep<<<NB * K2P, 256, 0, stream>>>(x, logits, wk, bk, kTg, vg);
  p2_q<<<NB * 64, 256, p2_lds, stream>>>(x, wq, bq, qg);
  p3_stats<<<NB * 16, 256, p3_lds, stream>>>(kTg, qg, pmax, psum);
  p3b_combine<<<(NB * K2P + 255) / 256, 256, 0, stream>>>(pmax, psum, smax, sinv);
  p4_fused<<<NB * HH, 256, p4_lds, stream>>>(kTg, vg, qg, logits, wo, bo, smax, sinv, out);
}
